// interaction_net_81896436400332
// MI455X (gfx1250) — matrix operand order repaired and checked by reading, not compiled, not run on hardware
//
#include <hip/hip_runtime.h>
#include <hip/hip_bf16.h>
#include <math.h>

typedef __attribute__((ext_vector_type(16))) __bf16 v16bf;
typedef __attribute__((ext_vector_type(8)))  float  v8f;

#define BB 32
#define TT 32
#define MM 4
#define CC 256
#define HH 256
#define LL 2
#define KK 6
#define G5H 1280
#define KT_W 8    // K tiles (K=256 / 32) for recurrence weights
#define NT_W 80   // N tiles (N=1280 / 16)
#define NQ   4    // N-tiles per wave in the fused gates GEMM

// ---------------------------------------------------------------------------
// Pack fp32 weight matrices [K][N] (row-major) into WMMA B-fragment layout:
// fragment element j of lane L (tile nt,kt):  K = kt*32 + (L>>4)*8 + j + (j>>3)*8,
// N = nt*16 + (L&15).  One contiguous 32B vector per lane per fragment.
// ---------------------------------------------------------------------------
__global__ void pack_w(const float* __restrict__ src, __bf16* __restrict__ dst,
                       int K, int N) {
  int KT = K >> 5, NT = N >> 4;
  int mat  = blockIdx.y;
  int tile = blockIdx.x;           // nt*KT + kt
  int nt = tile / KT, kt = tile % KT;
  int lane = threadIdx.x;
  const float* s = src + (size_t)mat * K * N;
  __bf16*      d = dst + (size_t)mat * KT * NT * 512;
  int n  = nt * 16 + (lane & 15);
  int kb = kt * 32 + ((lane >> 4) << 3);
  v16bf frag;
#pragma unroll
  for (int j = 0; j < 16; ++j) frag[j] = (__bf16)s[(size_t)(kb + j + ((j >> 3) << 3)) * N + n];
  *(v16bf*)(d + (((size_t)(nt * KT + kt)) * 32 + lane) * 16) = frag;
}

// ---------------------------------------------------------------------------
// Hard separation of the M=4 agents into 2 groups (softmax > 1/M threshold),
// producing layer-0 input seq [T][B][2][C].
// ---------------------------------------------------------------------------
__global__ void group_kernel(const float* __restrict__ feat,
                             const float* __restrict__ Wsep,
                             const float* __restrict__ bsep,
                             float* __restrict__ seq0) {
  int bt = blockIdx.x; int b = bt / TT, t = bt % TT;
  int c = threadIdx.x;
  __shared__ float red[256];
  __shared__ float sm[4];
  __shared__ float sel[4];
  const float* f = feat + (size_t)bt * MM * CC;
  for (int m = 0; m < MM; ++m) {
    red[c] = f[m * CC + c] * Wsep[c];
    __syncthreads();
    for (int s = 128; s > 0; s >>= 1) { if (c < s) red[c] += red[c + s]; __syncthreads(); }
    if (c == 0) sm[m] = red[0] + bsep[0];
    __syncthreads();
  }
  if (c == 0) {
    float mx = sm[0]; for (int m = 1; m < 4; ++m) mx = fmaxf(mx, sm[m]);
    float e[4], den = 0.f;
    for (int m = 0; m < 4; ++m) { e[m] = expf(sm[m] - mx); den += e[m]; }
    for (int m = 0; m < 4; ++m) sel[m] = ((e[m] / den - 0.25f) > 0.f) ? 1.f : 0.f;
  }
  __syncthreads();
  float s0 = 0.f, s1 = 0.f, d0 = 0.f, d1 = 0.f;
  for (int m = 0; m < 4; ++m) {
    float v = f[m * CC + c];
    s0 += sel[m] * v;          s1 += (1.f - sel[m]) * v;
    d0 += sel[m];              d1 += 1.f - sel[m];
  }
  size_t base = ((size_t)t * BB + b) * 2 * CC + c;
  seq0[base]      = s0 / (d0 + 1e-8f);
  seq0[base + CC] = s1 / (d1 + 1e-8f);
}

// ---------------------------------------------------------------------------
// share_feat = relu(feat.reshape(B*T, 1024) @ W_l2s + b) via WMMA bf16.
// Grid: (64 M-tiles, 16 N-tiles), one wave per block, K loop of 32.
// ---------------------------------------------------------------------------
__global__ void sharefeat_kernel(const float* __restrict__ feat,
                                 const __bf16* __restrict__ wp,
                                 const float* __restrict__ bias,
                                 float* __restrict__ sf) {
  const int K = 1024, KT = 32;
  int mt = blockIdx.x, nt = blockIdx.y;
  int lane = threadIdx.x;
  int row  = mt * 16 + (lane & 15);
  int kb0  = ((lane >> 4) << 3);
  v8f acc = {};
  for (int kt = 0; kt < KT; ++kt) {
    v16bf a;
#pragma unroll
    for (int j = 0; j < 8; ++j) {
      int cb = kt * 32 + kb0;
      a[j]     = (__bf16)feat[(size_t)row * K + cb + j];
      a[j + 8] = (__bf16)feat[(size_t)row * K + cb + 16 + j];
    }
    v16bf bf = *(const v16bf*)(wp + (((size_t)(nt * KT + kt)) * 32 + lane) * 16);
    acc = __builtin_amdgcn_wmma_f32_16x16x32_bf16(false, a, false, bf,
                                                  (short)0, acc, false, false);
  }
  int n = nt * 16 + (lane & 15);
#pragma unroll
  for (int r = 0; r < 8; ++r) {
    int m = mt * 16 + r + ((lane >> 4) << 3);
    sf[(size_t)m * HH + n] = fmaxf(acc[r] + bias[n], 0.f);
  }
}

// ---------------------------------------------------------------------------
// Standalone routing, used only at t==0 of each layer (h_{-1}=0 -> logits =
// b_arch, hard top-2 with first-index tie-breaking).  One wave per (b,g).
// ---------------------------------------------------------------------------
__global__ void route_kernel(const float* __restrict__ barch,
                             float* __restrict__ wroute, int l) {
  int bg = blockIdx.x;
  int lane = threadIdx.x;
  if (lane == 0) {
    float logit[KK];
    for (int k = 0; k < KK; ++k) logit[k] = barch[l * KK + k];
    int i1 = 0;
    for (int k = 1; k < KK; ++k) if (logit[k] > logit[i1]) i1 = k;
    int i2 = (i1 == 0) ? 1 : 0;
    for (int k = 0; k < KK; ++k) if (k != i1 && logit[k] > logit[i2]) i2 = k;
    for (int k = 0; k < KK; ++k)
      wroute[bg * KK + k] = (k == i1 || k == i2) ? 0.5f : 0.f;
  }
}

// ---------------------------------------------------------------------------
// Fused gates GEMM with in-register A-fragment build (routing weight folded
// into candidate rows) and 4 N-tile accumulators per wave:
//   gates = x@Wx + h@Wh + sum_k w_k*(cand_k@Wc_k) + bx + bh + sum_k w_k*bc_k
// Grid: (20 N-groups, 4 = g*2+mt), one wave/block, 256 v_wmma per wave.
// A fragments built once per (i,kt) and reused for 4 WMMAs (4x A-reuse).
// ---------------------------------------------------------------------------
__global__ void gates_fused(const float* __restrict__ seq_in,
                            const float* __restrict__ hs_out,
                            const float* __restrict__ context,
                            const float* __restrict__ wroute,
                            const __bf16* __restrict__ wxp,
                            const __bf16* __restrict__ whp,
                            const __bf16* __restrict__ wcp,
                            const float* __restrict__ bx,
                            const float* __restrict__ bh,
                            const float* __restrict__ bc,
                            float* __restrict__ gates, int l, int t) {
  int ntg = blockIdx.x;            // group of NQ consecutive n-tiles
  int gm  = blockIdx.y;            // g*2 + mt
  int g = gm >> 1, mt = gm & 1;
  int lane = threadIdx.x;
  int brow  = mt * 16 + (lane & 15);   // matrix row held by this lane (A layout)
  int khalf = ((lane >> 4) << 3);      // K sub-offset per ISA A-fragment map

  const float* w = wroute + (brow * 2 + g) * KK;
  const float* xt  = seq_in + (((size_t)t * BB + brow) * 2 + g) * CC;
  const float* xts = seq_in + (((size_t)t * BB + brow) * 2 + (g ^ 1)) * CC;
  const float* xn  = (t + 1 < TT) ? seq_in + (((size_t)(t + 1) * BB + brow) * 2 + g) * CC : nullptr;
  const float* xns = (t + 1 < TT) ? seq_in + (((size_t)(t + 1) * BB + brow) * 2 + (g ^ 1)) * CC : nullptr;
  const float* h1  = (t >= 1) ? hs_out + (((size_t)(t - 1) * BB + brow) * 2 + g) * HH : nullptr;
  const float* h1s = (t >= 1) ? hs_out + (((size_t)(t - 1) * BB + brow) * 2 + (g ^ 1)) * HH : nullptr;
  const float* h2  = (t >= 2) ? hs_out + (((size_t)(t - 2) * BB + brow) * 2 + g) * HH : nullptr;
  const float* sc  = context + ((size_t)brow * TT + t) * HH;
  const float* srcs[8]  = { xt, h1, h2, xn, h1s, xts, xns, sc };
  float        alpha[8] = { 1.f, 1.f, w[0], w[1], w[2], w[3], w[4], w[5] };

  const size_t WS = (size_t)KT_W * NT_W * 512;   // elems per packed matrix
  int lg = l * 2 + g;
  const __bf16* wptr[8];
  wptr[0] = wxp + (size_t)lg * WS;
  wptr[1] = whp + (size_t)lg * WS;
#pragma unroll
  for (int k = 0; k < 6; ++k) wptr[2 + k] = wcp + ((size_t)lg * 6 + k) * WS;

  v8f acc[NQ];
#pragma unroll
  for (int q = 0; q < NQ; ++q) acc[q] = (v8f){};

#pragma unroll
  for (int i = 0; i < 8; ++i) {
    const float* s = srcs[i];
    float al = alpha[i];
    const __bf16* wp = wptr[i];
    for (int kt = 0; kt < KT_W; ++kt) {
      v16bf a;
      int cb = kt * 32 + khalf;
#pragma unroll
      for (int j = 0; j < 8; ++j) {           // 2x contiguous 32B fp32 runs
        a[j]     = (__bf16)(s ? al * s[cb + j] : 0.f);
        a[j + 8] = (__bf16)(s ? al * s[cb + 16 + j] : 0.f);
      }
      const __bf16* wb = wp + (((size_t)(ntg * NQ) * KT_W + kt) * 32 + lane) * 16;
#pragma unroll
      for (int q = 0; q < NQ; ++q) {
        v16bf bf = *(const v16bf*)(wb + (size_t)q * KT_W * 512);
        acc[q] = __builtin_amdgcn_wmma_f32_16x16x32_bf16(false, a, false, bf,
                                                         (short)0, acc[q], false, false);
      }
    }
  }

#pragma unroll
  for (int q = 0; q < NQ; ++q) {
    int n = (ntg * NQ + q) * 16 + (lane & 15);
    float bias0 = bx[(size_t)lg * G5H + n] + bh[(size_t)lg * G5H + n];
#pragma unroll
    for (int r = 0; r < 8; ++r) {
      int b2 = mt * 16 + r + ((lane >> 4) << 3);     // C/D layout row
      const float* wr = wroute + (b2 * 2 + g) * KK;
      float bias = bias0;
#pragma unroll
      for (int k = 0; k < 6; ++k)
        bias += wr[k] * bc[((size_t)lg * KK + k) * G5H + n];
      gates[((size_t)g * BB + b2) * G5H + n] = acc[q][r] + bias;
    }
  }
}

// ---------------------------------------------------------------------------
// Pointwise LSTM update + shared-state accumulation + output write, fused with
// next step's routing: block b holds all 256 h_new values, so h@W_arch is an
// LDS tree-reduce; hard top-2 -> w_{t+1} in {0, 0.5}.
// ---------------------------------------------------------------------------
__global__ void update_route(const float* __restrict__ gates,
                             const float* __restrict__ sf,
                             const float* __restrict__ Warch,
                             const float* __restrict__ barch,
                             float* __restrict__ cell,
                             float* __restrict__ sstate,
                             float* __restrict__ cgate,
                             float* __restrict__ hs_out,
                             float* __restrict__ wroute,
                             float* __restrict__ out, int l, int t) {
  int b = blockIdx.x;
  int h = threadIdx.x;
  float sgsum = 0.f;
  float hnew[2];
#pragma unroll
  for (int g = 0; g < 2; ++g) {
    const float* gt = gates + ((size_t)g * BB + b) * G5H;
    float ig = gt[h], fg = gt[256 + h], og = gt[512 + h], sg = gt[768 + h], cc = gt[1024 + h];
    float cprev = (t > 0) ? cell[(b * 2 + g) * HH + h] : 0.f;
    float si  = 1.f / (1.f + expf(-ig));
    float sfg = 1.f / (1.f + expf(-fg));
    float cnew = (1.f - sfg) * cprev + si * tanhf(cc);
    float hn = (1.f / (1.f + expf(-og))) * tanhf(cnew);
    cell[(b * 2 + g) * HH + h] = cnew;
    hs_out[(((size_t)t * BB + b) * 2 + g) * HH + h] = hn;
    hnew[g] = hn;
    sgsum += sg;
  }
  float obj = 1.f / (1.f + expf(-sgsum));
  float ss = ((t > 0) ? sstate[b * HH + h] : 0.f) + obj * sf[((size_t)b * TT + t) * HH + h];
  float cg = ((t > 0) ? cgate[b * HH + h] : 0.f) + obj;
  sstate[b * HH + h] = ss;
  cgate[b * HH + h] = cg;
  float* o = out + (((size_t)l * BB + b) * TT + t) * (3 * HH);
  o[h]          = hnew[0];
  o[HH + h]     = hnew[1];
  o[2 * HH + h] = ss / cg;

  // ---- routing for step t+1: logits[g][k] = sum_h hnew[g][h]*Warch[l][h][k]
  __shared__ float red[256];
  __shared__ float lg[2 * KK];
  for (int gk = 0; gk < 2 * KK; ++gk) {
    int g = gk / KK, k = gk % KK;
    red[h] = hnew[g] * Warch[((size_t)l * HH + h) * KK + k];
    __syncthreads();
    for (int s = 128; s > 0; s >>= 1) { if (h < s) red[h] += red[h + s]; __syncthreads(); }
    if (h == 0) lg[gk] = red[0] + barch[l * KK + k];
    __syncthreads();
  }
  if (h < 2) {
    const float* lgg = lg + h * KK;
    int i1 = 0;
    for (int k = 1; k < KK; ++k) if (lgg[k] > lgg[i1]) i1 = k;
    int i2 = (i1 == 0) ? 1 : 0;
    for (int k = 0; k < KK; ++k) if (k != i1 && lgg[k] > lgg[i2]) i2 = k;
    for (int k = 0; k < KK; ++k)
      wroute[(b * 2 + h) * KK + k] = (k == i1 || k == i2) ? 0.5f : 0.f;
  }
}

// ---------------------------------------------------------------------------
extern "C" void kernel_launch(void* const* d_in, const int* in_sizes, int n_in,
                              void* d_out, int out_size, void* d_ws, size_t ws_size,
                              hipStream_t stream) {
  const float* feat  = (const float*)d_in[0];
  const float* ctx   = (const float*)d_in[1];
  const float* Wl2s  = (const float*)d_in[2];
  const float* bl2s  = (const float*)d_in[3];
  const float* Wsep  = (const float*)d_in[4];
  const float* bsep  = (const float*)d_in[5];
  const float* Warch = (const float*)d_in[6];
  const float* barch = (const float*)d_in[7];
  const float* Wx    = (const float*)d_in[8];
  const float* bx    = (const float*)d_in[9];
  const float* Wh    = (const float*)d_in[10];
  const float* bh    = (const float*)d_in[11];
  const float* Wc    = (const float*)d_in[12];
  const float* bc    = (const float*)d_in[13];
  float* out = (float*)d_out;

  char* ws = (char*)d_ws;
  size_t off = 0;
  auto carve = [&](size_t bytes) -> void* {
    void* p = ws + off;
    off = (off + bytes + 255) & ~(size_t)255;
    return p;
  };
  __bf16* wl2s_p = (__bf16*)carve((size_t)1024 * 256 * 2);        // packed W_l2s
  __bf16* wx_p   = (__bf16*)carve((size_t)4 * 256 * 1280 * 2);    // packed W_x
  __bf16* wh_p   = (__bf16*)carve((size_t)4 * 256 * 1280 * 2);    // packed W_h
  __bf16* wc_p   = (__bf16*)carve((size_t)24 * 256 * 1280 * 2);   // packed W_can
  float*  seqA   = (float*)carve((size_t)TT * BB * 2 * CC * 4);
  float*  seqB   = (float*)carve((size_t)TT * BB * 2 * CC * 4);
  float*  sf     = (float*)carve((size_t)BB * TT * HH * 4);
  float*  gates  = (float*)carve((size_t)2 * BB * G5H * 4);
  float*  wroute = (float*)carve((size_t)BB * 2 * KK * 4);
  float*  cell   = (float*)carve((size_t)BB * 2 * HH * 4);
  float*  sstate = (float*)carve((size_t)BB * HH * 4);
  float*  cgate  = (float*)carve((size_t)BB * HH * 4);

  // one-time weight packing (deterministic, re-done every call)
  pack_w<<<dim3(16 * 32, 1), 32, 0, stream>>>(Wl2s, wl2s_p, 1024, 256);
  pack_w<<<dim3(NT_W * KT_W, 4), 32, 0, stream>>>(Wx, wx_p, 256, 1280);
  pack_w<<<dim3(NT_W * KT_W, 4), 32, 0, stream>>>(Wh, wh_p, 256, 1280);
  pack_w<<<dim3(NT_W * KT_W, 24), 32, 0, stream>>>(Wc, wc_p, 256, 1280);

  group_kernel<<<BB * TT, 256, 0, stream>>>(feat, Wsep, bsep, seqA);
  sharefeat_kernel<<<dim3(64, 16), 32, 0, stream>>>(feat, wl2s_p, bl2s, sf);

  float* seq_in  = seqA;   // layer input sequence [T][B][2][C]
  float* seq_out = seqB;   // layer hidden outputs [T][B][2][H]
  for (int l = 0; l < LL; ++l) {
    route_kernel<<<BB * 2, 32, 0, stream>>>(barch, wroute, l);   // t=0: h=0
    for (int t = 0; t < TT; ++t) {
      gates_fused<<<dim3(NT_W / NQ, 4), 32, 0, stream>>>(
          seq_in, seq_out, ctx, wroute, wx_p, wh_p, wc_p, bx, bh, bc, gates, l, t);
      update_route<<<BB, 256, 0, stream>>>(gates, sf, Warch, barch, cell,
                                           sstate, cgate, seq_out, wroute, out, l, t);
    }
    float* tmp = seq_in; seq_in = seq_out; seq_out = tmp;
  }
}
